// WindowAttention_36404142801256
// MI455X (gfx1250) — matrix operand order repaired and checked by reading, not compiled, not run on hardware
//
#include <hip/hip_runtime.h>
#include <hip/hip_bf16.h>
#include <math.h>
#include <stdint.h>

// ---------------------------------------------------------------------------
// x: (B=4, T=8, H=20, W=36, C=512); windows Nw=720, top-k 180
// GEMM1: (5760 x 512) @ (512 x 6272)
// fold -> (32,128,30,54); up2 -> (32,128,60,108); two 3x3 convs 128->128
// GEMM2 (implicit unfold): (23040 x 6272) @ (6272 x 512) + residual
// ---------------------------------------------------------------------------
static constexpr int CB = 4, CT = 8, CH_ = 20, CW_ = 36, CC = 512;
static constexpr int NW = 720, NSEL = 180;
static constexpr int W2 = 18;
static constexpr int BT = 32;
static constexpr int CHN = 128;
static constexpr int OH = 30, OW = 54;
static constexpr int UH = 60, UW = 108;
static constexpr int CKK = 6272;
static constexpr int KCONV = CHN * 9;           // 1152
static constexpr int LDSP = 68;                 // A-tile LDS row pitch

typedef __bf16 bf16_t;
typedef __attribute__((ext_vector_type(16))) bf16_t v16bf;
typedef __attribute__((ext_vector_type(8)))  float  v8f;
typedef __attribute__((ext_vector_type(4)))  unsigned int u32x4;
typedef __attribute__((ext_vector_type(8)))  int i32x8;
typedef __attribute__((ext_vector_type(4)))  int i32x4;

#define WMMA_BF16(A_, B_, C_) \
  __builtin_amdgcn_wmma_f32_16x16x32_bf16(false, (A_), false, (B_), (short)0, (C_), false, false)

// Tensor Data Mover availability (clang-22: 5-arg builtin; clang-23+: 6-arg)
#if defined(__has_builtin)
#  if __has_builtin(__builtin_amdgcn_tensor_load_to_lds) && \
      __has_builtin(__builtin_amdgcn_s_wait_tensorcnt)
#    define HAVE_TDM 1
#  else
#    define HAVE_TDM 0
#  endif
#else
#  define HAVE_TDM 0
#endif

// Async 16B global->LDS copy (GLOBAL_LOAD_ASYNC_TO_LDS_B128, GV mode,
// ASYNCcnt-tracked). ldsaddr = wave-relative LDS byte address.
__device__ __forceinline__ void async_g2l_b128(unsigned ldsaddr, const float* gsrc) {
  asm volatile("global_load_async_to_lds_b128 %0, %1, off"
               :: "v"(ldsaddr), "v"(gsrc) : "memory");
}
__device__ __forceinline__ void wait_async0() {
  asm volatile("s_wait_asynccnt 0x0" ::: "memory");
}

// TDM: load a 2D f32 tile (tile_x contiguous elems x tile_y rows, row stride
// stride_elems) from global into LDS at lds_addr.  Descriptor layout per
// CDNA5 ISA.  Optional LDS padding (load only):
// pad_amount+1 DWORDs inserted every 2<<pad_interval DWORDs.
__device__ __forceinline__ void tdm_load_2d_f32(unsigned lds_addr, const float* g,
                                                int tile_x, int tile_y,
                                                int stride_elems, int tens_x, int tens_y,
                                                int pad_en, int pad_interval, int pad_amount) {
#if HAVE_TDM
  const unsigned long long ga = (unsigned long long)(uintptr_t)g;
  u32x4 g0 = {
    1u,                                               // count=1 (valid user D#)
    lds_addr,                                         // lds_addr (bytes)
    (unsigned)(ga & 0xFFFFFFFFu),                     // global_addr[31:0]
    (unsigned)((ga >> 32) & 0x01FFFFFFu) | 0x80000000u // global_addr[56:32] | type=2
  };
  i32x8 g1 = {
    (int)(0x20000u | ((unsigned)pad_en << 20) |       // data_size=2 (4B) + padding
          ((unsigned)pad_interval << 22) | ((unsigned)pad_amount << 25)),
    (int)(((unsigned)tens_x & 0xFFFFu) << 16),        // [63:48] tensor_dim0 lo16
    (int)((((unsigned)tens_x >> 16) & 0xFFFFu) |      // tensor_dim0 hi16
          (((unsigned)tens_y & 0xFFFFu) << 16)),      // tensor_dim1 lo16
    (int)((((unsigned)tens_y >> 16) & 0xFFFFu) |      // tensor_dim1 hi16
          ((unsigned)tile_x << 16)),                  // [127:112] tile_dim0
    (int)((unsigned)tile_y & 0xFFFFu),                // tile_dim1 (tile_dim2=0)
    stride_elems,                                     // tensor_dim0_stride[31:0]
    0, 0
  };
  i32x4 z4 = {0, 0, 0, 0};
#if __clang_major__ >= 23
  i32x8 z8 = {0, 0, 0, 0, 0, 0, 0, 0};
  __builtin_amdgcn_tensor_load_to_lds(g0, g1, z4, z4, z8, 0);
#else
  __builtin_amdgcn_tensor_load_to_lds(g0, g1, z4, z4, 0);
#endif
#else
  (void)lds_addr; (void)g; (void)tile_x; (void)tile_y; (void)stride_elems;
  (void)tens_x; (void)tens_y; (void)pad_en; (void)pad_interval; (void)pad_amount;
#endif
}
__device__ __forceinline__ void wait_tensor0() {
#if HAVE_TDM
  __builtin_amdgcn_s_wait_tensorcnt(0);
#endif
}

// ---------------------------------------------------------------------------
// Wave32 WMMA fragment helpers (CDNA5 ISA §7.12.2)
// ---------------------------------------------------------------------------
template <typename FA>
__device__ __forceinline__ v16bf load_a_frag(FA f, int k0) {
  const int lane = threadIdx.x & 31;
  const int m    = lane & 15;
  const int kb   = (lane >> 4) ? 16 : 0;
  v16bf a;
#pragma unroll
  for (int h = 0; h < 16; ++h) a[h] = (bf16_t)f(m, k0 + kb + h);
  return a;
}

template <typename FB>
__device__ __forceinline__ v16bf load_b_frag(FB f, int k0) {
  const int lane = threadIdx.x & 31;
  const int n    = lane & 15;
  const int kb   = (lane >> 4) ? 16 : 0;
  v16bf b;
#pragma unroll
  for (int h = 0; h < 16; ++h) b[h] = (bf16_t)f(k0 + kb + h, n);
  return b;
}

template <typename FS>
__device__ __forceinline__ void store_d_frag(const v8f& d, FS f) {
  const int lane = threadIdx.x & 31;
  const int n    = lane & 15;
  const int mb   = (lane >> 4) ? 8 : 0;
#pragma unroll
  for (int r = 0; r < 8; ++r) f(mb + r, n, d[r]);
}

// ---------------------------------------------------------------------------
// 1) Window means
// ---------------------------------------------------------------------------
__global__ void k_means(const float* __restrict__ x, float* __restrict__ means) {
  const int w  = blockIdx.x;
  const int b  = w / 180;
  const int wh = (w / 18) % 10;
  const int ww = w % 18;
  for (int c = threadIdx.x; c < CC; c += blockDim.x) {
    float s = 0.f;
#pragma unroll
    for (int t = 0; t < CT; ++t)
      for (int a = 0; a < 2; ++a)
        for (int d = 0; d < 2; ++d)
          s += x[((((size_t)b * CT + t) * CH_ + (wh * 2 + a)) * CW_ + (ww * 2 + d)) * CC + c];
    means[(size_t)w * CC + c] = s * (1.f / 32.f);
  }
}

// ---------------------------------------------------------------------------
// Generic pipelined GEMM: D(MxN) = A(MxK)*B(KxN) + bias.
// grid = (M/MTILE, N/128); block = 256 (8 waves; wave -> one 16-col N tile).
// A: double-buffered async b128 DMA (ASYNCcnt). B: double-buffered TDM 2D
// tile DMA (TENSORcnt), issued by wave 0.  One barrier per 64-K chunk;
// chunk i+1 DMA overlaps chunk i WMMA.
// ---------------------------------------------------------------------------
template <int MTILE, int KSZ, int LDA, int LDB, int LDD>
__global__ void k_gemm_ws(const float* __restrict__ A, const float* __restrict__ B,
                          const float* __restrict__ bias, float* __restrict__ D) {
  constexpr int MH  = MTILE / 16;
  constexpr int NCH = KSZ / 64;
  __shared__ __align__(16) float As[2][MTILE][LDSP];
#if HAVE_TDM
  __shared__ __align__(16) float Bs[2][64][128];
#endif
  const int mt    = blockIdx.x * MTILE;
  const int wave  = threadIdx.x >> 5;
  const int nb0   = blockIdx.y * 128;
  const int nbase = nb0 + wave * 16;
  v8f acc[MH] = {};

  auto issueA = [&](int ch) {
    const int k0 = ch * 64;
#pragma unroll
    for (int i = 0; i < MH; ++i) {
      const int g   = threadIdx.x + i * 256;
      const int row = g >> 4;
      const int c4  = (g & 15) * 4;
      async_g2l_b128((unsigned)(uintptr_t)&As[ch & 1][row][c4],
                     &A[(size_t)(mt + row) * LDA + k0 + c4]);
    }
  };
  auto issueB = [&](int ch) {
#if HAVE_TDM
    if (wave == 0)
      tdm_load_2d_f32((unsigned)(uintptr_t)&Bs[ch & 1][0][0],
                      &B[(size_t)(ch * 64) * LDB + nb0],
                      /*tile*/128, 64, /*stride*/LDB, /*tensor*/LDB, KSZ - ch * 64,
                      0, 0, 0);
#endif
  };

  issueA(0); issueB(0);
  for (int ch = 0; ch < NCH; ++ch) {
    wait_async0();
    if (wave == 0) wait_tensor0();
    __syncthreads();
    if (ch + 1 < NCH) { issueA(ch + 1); issueB(ch + 1); }
#if !HAVE_TDM
    const int k0g = ch * 64;
    if (ch + 1 < NCH) __builtin_prefetch(&B[(size_t)(k0g + 64) * LDB + nbase], 0, 1);
#endif
#pragma unroll
    for (int kc = 0; kc < 2; ++kc) {
      const int kk = kc * 32;
#if HAVE_TDM
      v16bf bfr = load_b_frag([&](int kx, int n) {
        return Bs[ch & 1][kx][wave * 16 + n];
      }, kk);
#else
      v16bf bfr = load_b_frag([&](int kx, int n) {
        return B[(size_t)(ch * 64 + kx) * LDB + nbase + n];
      }, kk);
#endif
#pragma unroll
      for (int h = 0; h < MH; ++h) {
        v16bf af = load_a_frag([&](int m, int kx) { return As[ch & 1][h * 16 + m][kx]; }, kk);
        acc[h] = WMMA_BF16(af, bfr, acc[h]);
      }
    }
  }
#pragma unroll
  for (int h = 0; h < MH; ++h)
    store_d_frag(acc[h], [&](int m, int n, float v) {
      D[(size_t)(mt + h * 16 + m) * LDD + nbase + n] = v + bias[nbase + n];
    });
}

// ---------------------------------------------------------------------------
// 2) Attention logits S = q @ k^T (720x720); one wave per 16x16 tile.
// ---------------------------------------------------------------------------
__global__ void k_attn(const float* __restrict__ q, const float* __restrict__ kmat,
                       float* __restrict__ S) {
  const int mt = blockIdx.x * 16, nt = blockIdx.y * 16;
  v8f acc = {};
  for (int k0 = 0; k0 < CC; k0 += 32) {
    v16bf af  = load_a_frag([&](int m, int kx) { return q[(size_t)(mt + m) * CC + kx]; }, k0);
    v16bf bfr = load_b_frag([&](int kx, int n) { return kmat[(size_t)(nt + n) * CC + kx]; }, k0);
    acc = WMMA_BF16(af, bfr, acc);
  }
  store_d_frag(acc, [&](int m, int n, float v) { S[(size_t)(mt + m) * NW + nt + n] = v; });
}

// ---------------------------------------------------------------------------
// 3) scores[i] = sum_j softmax(S[i,:])[j]
// ---------------------------------------------------------------------------
__global__ void k_scores(const float* __restrict__ S, float* __restrict__ scores) {
  __shared__ float red[256];
  const int tid = threadIdx.x;
  const float* row = S + (size_t)blockIdx.x * NW;

  float mx = -INFINITY;
  for (int j = tid; j < NW; j += 256) mx = fmaxf(mx, row[j]);
  red[tid] = mx; __syncthreads();
  for (int s = 128; s > 0; s >>= 1) {
    if (tid < s) red[tid] = fmaxf(red[tid], red[tid + s]);
    __syncthreads();
  }
  mx = red[0]; __syncthreads();

  float se = 0.f;
  for (int j = tid; j < NW; j += 256) se += __expf(row[j] - mx);
  red[tid] = se; __syncthreads();
  for (int s = 128; s > 0; s >>= 1) {
    if (tid < s) red[tid] += red[tid + s];
    __syncthreads();
  }
  const float total = red[0]; __syncthreads();

  float sc = 0.f;
  for (int j = tid; j < NW; j += 256) sc += __expf(row[j] - mx) / total;
  red[tid] = sc; __syncthreads();
  for (int s = 128; s > 0; s >>= 1) {
    if (tid < s) red[tid] += red[tid + s];
    __syncthreads();
  }
  if (tid == 0) scores[blockIdx.x] = red[0];
}

// ---------------------------------------------------------------------------
// 4) Stable top-k (k=180 of 720), ties -> lowest index.
// ---------------------------------------------------------------------------
__global__ void k_topk(const float* __restrict__ scores, int* __restrict__ idx) {
  __shared__ float bv[256];
  __shared__ int   bi[256];
  __shared__ unsigned char taken[NW];
  const int tid = threadIdx.x;
  for (int j = tid; j < NW; j += 256) taken[j] = 0;
  __syncthreads();
  for (int it = 0; it < NSEL; ++it) {
    float best = -INFINITY; int bidx = NW;
    for (int j = tid; j < NW; j += 256) {
      if (!taken[j]) {
        float v = scores[j];
        if (v > best || (v == best && j < bidx)) { best = v; bidx = j; }
      }
    }
    bv[tid] = best; bi[tid] = bidx; __syncthreads();
    for (int s = 128; s > 0; s >>= 1) {
      if (tid < s) {
        if (bv[tid + s] > bv[tid] || (bv[tid + s] == bv[tid] && bi[tid + s] < bi[tid])) {
          bv[tid] = bv[tid + s]; bi[tid] = bi[tid + s];
        }
      }
      __syncthreads();
    }
    if (tid == 0) { idx[it] = bi[0]; taken[bi[0]] = 1; }
    __syncthreads();
  }
}

// ---------------------------------------------------------------------------
// 5) Gather selected windows -> Y (5760 x 512)
// ---------------------------------------------------------------------------
__global__ void k_gather(const float* __restrict__ x, const int* __restrict__ idx,
                         float* __restrict__ Y) {
  const int g  = blockIdx.x;
  const int b  = g / 1440;
  const int r  = g % 1440;
  const int t  = r / 180;
  const int l  = r % 180;
  const int h2 = l / W2, w2 = l % W2;
  const int i = h2 >> 1, a = h2 & 1, j = w2 >> 1, bb = w2 & 1;
  const int widx = idx[b * 45 + i * 9 + j];
  const int bw = widx / 180, wh = (widx / 18) % 10, ww = widx % 18;
  const float4* src = (const float4*)(x +
      ((((size_t)bw * CT + t) * CH_ + (wh * 2 + a)) * CW_ + (ww * 2 + bb)) * CC);
  float4* dst = (float4*)(Y + (size_t)g * CC);
  dst[threadIdx.x] = src[threadIdx.x];              // 128 threads x float4
}

// ---------------------------------------------------------------------------
// 6) fold (gather form) + sc_bias
// ---------------------------------------------------------------------------
__global__ void k_fold(const float* __restrict__ feat, const float* __restrict__ sc_bias,
                       float* __restrict__ out) {
  const int id = blockIdx.x * 256 + threadIdx.x;
  if (id >= BT * CHN * OH * OW) return;
  const int xx = id % OW;
  const int yy = (id / OW) % OH;
  const int c  = (id / (OW * OH)) % CHN;
  const int n  = id / (OW * OH * CHN);
  const int b = n / CT, t = n % CT;
  float acc = sc_bias[(size_t)c * OH * OW + yy * OW + xx];
#pragma unroll
  for (int kh = 0; kh < 7; ++kh) {
    const int ty = yy + 3 - kh;
    if (ty % 3 != 0) continue;
    const int ph = ty / 3;
    if ((unsigned)ph >= 10u) continue;
#pragma unroll
    for (int kw = 0; kw < 7; ++kw) {
      const int tx = xx + 3 - kw;
      if (tx % 3 != 0) continue;
      const int pw = tx / 3;
      if ((unsigned)pw >= 18u) continue;
      const int g = b * 1440 + t * 180 + ph * 18 + pw;
      acc += feat[(size_t)g * CKK + c * 49 + kh * 7 + kw];
    }
  }
  out[id] = acc;
}

// ---------------------------------------------------------------------------
// 7) bilinear x2, align_corners
// ---------------------------------------------------------------------------
__global__ void k_up2(const float* __restrict__ in, float* __restrict__ out) {
  const int id = blockIdx.x * 256 + threadIdx.x;
  if (id >= BT * CHN * UH * UW) return;
  const int xx = id % UW;
  const int yy = (id / UW) % UH;
  const int nc = id / (UW * UH);
  const float sy = yy * (29.f / 59.f);
  const float sx = xx * (53.f / 107.f);
  int y0 = (int)floorf(sy); int y1 = y0 + 1; if (y1 > OH - 1) y1 = OH - 1;
  int x0 = (int)floorf(sx); int x1 = x0 + 1; if (x1 > OW - 1) x1 = OW - 1;
  const float fy = sy - (float)y0, fx = sx - (float)x0;
  const float* p = in + (size_t)nc * OH * OW;
  out[id] = (1.f - fy) * ((1.f - fx) * p[y0 * OW + x0] + fx * p[y0 * OW + x1]) +
            fy         * ((1.f - fx) * p[y1 * OW + x0] + fx * p[y1 * OW + x1]);
}

// ---------------------------------------------------------------------------
// 8) implicit-GEMM 3x3 conv 128->128 (+optional leaky 0.2).
//    M-tile 32 pixels, N = 128 (8 waves x 16), K = 1152 in chunks of 64.
//    A: zero-padded im2col gathered to double-buffered LDS.
//    B: conv weights [co][k] tile via TDM with LDS padding (pitch 68 floats)
//       to avoid bank conflicts on the k-major reads.
// ---------------------------------------------------------------------------
__global__ void k_conv3(const float* __restrict__ in, const float* __restrict__ w,
                        const float* __restrict__ bias, float* __restrict__ out,
                        int leaky) {
  constexpr int NCH = KCONV / 64;
  __shared__ __align__(16) float As[2][32][LDSP];
#if HAVE_TDM
  __shared__ __align__(16) float Bs[2][128][68];   // pitch 68 via TDM pad
#endif
  const int mt   = blockIdx.x * 32;
  const int wave = threadIdx.x >> 5;
  v8f acc[2] = {};

  auto stageA = [&](int ch) {
    const int k0 = ch * 64;
#pragma unroll
    for (int i = 0; i < 8; ++i) {
      const int e   = threadIdx.x + i * 256;     // 32 rows x 64 k
      const int row = e >> 6;
      const int kx  = (e & 63) + k0;
      const int gm  = mt + row;
      const int n   = gm / (UH * UW);
      const int pix = gm % (UH * UW);
      const int yy = pix / UW, xx = pix % UW;
      const int ci = kx / 9, rr = kx % 9;
      const int iy = yy + rr / 3 - 1, ix = xx + rr % 3 - 1;
      float v = 0.f;
      if ((unsigned)iy < (unsigned)UH && (unsigned)ix < (unsigned)UW)
        v = in[(((size_t)n * CHN + ci) * UH + iy) * UW + ix];
      As[ch & 1][row][e & 63] = v;
    }
  };
  auto issueB = [&](int ch) {
#if HAVE_TDM
    if (wave == 0)
      tdm_load_2d_f32((unsigned)(uintptr_t)&Bs[ch & 1][0][0],
                      &w[ch * 64],
                      /*tile*/64, 128, /*stride*/KCONV, /*tensor*/KCONV, CHN,
                      /*pad*/1, /*every 64 dw*/5, /*4 dw*/3);
#endif
  };

  stageA(0); issueB(0);
  for (int ch = 0; ch < NCH; ++ch) {
    if (wave == 0) wait_tensor0();
    __syncthreads();
    if (ch + 1 < NCH) { issueB(ch + 1); stageA(ch + 1); }
#pragma unroll
    for (int kc = 0; kc < 2; ++kc) {
      const int kk = kc * 32;
#if HAVE_TDM
      v16bf bfr = load_b_frag([&](int kx, int n) {
        return Bs[ch & 1][wave * 16 + n][kx];
      }, kk);
#else
      v16bf bfr = load_b_frag([&](int kx, int n) {
        return w[(size_t)(wave * 16 + n) * KCONV + ch * 64 + kx];
      }, kk);
#endif
#pragma unroll
      for (int h = 0; h < 2; ++h) {
        v16bf af = load_a_frag([&](int m, int kx) { return As[ch & 1][h * 16 + m][kx]; }, kk);
        acc[h] = WMMA_BF16(af, bfr, acc[h]);
      }
    }
  }
#pragma unroll
  for (int h = 0; h < 2; ++h)
    store_d_frag(acc[h], [&](int m, int n, float v) {
      const int gm = mt + h * 16 + m;
      const int co = wave * 16 + n;
      const int nn = gm / (UH * UW), pix = gm % (UH * UW);
      float r = v + bias[co];
      if (leaky) r = r > 0.f ? r : 0.2f * r;
      out[((size_t)nn * CHN + co) * (UH * UW) + pix] = r;
    });
}

// ---------------------------------------------------------------------------
// 9) implicit unfold + GEMM2 + residual -> out
//    M-tile 32 rows, N = 512 (grid.y*128 + wave*16), K = 6272 chunks of 64.
//    A: unfold gather, double-buffered LDS.  B: ss_w tile via TDM.
// ---------------------------------------------------------------------------
__global__ void k_gemm2(const float* __restrict__ d2, const float* __restrict__ ss_w,
                        const float* __restrict__ ss_b, const float* __restrict__ x,
                        float* __restrict__ out) {
  constexpr int NCH = CKK / 64;
  __shared__ __align__(16) float As[2][32][LDSP];
#if HAVE_TDM
  __shared__ __align__(16) float Bs[2][64][128];
#endif
  const int mt    = blockIdx.x * 32;
  const int wave  = threadIdx.x >> 5;
  const int nb0   = blockIdx.y * 128;
  const int nbase = nb0 + wave * 16;
  v8f acc[2] = {};

  auto stageA = [&](int ch) {
    const int k0 = ch * 64;
#pragma unroll
    for (int i = 0; i < 8; ++i) {
      const int e   = threadIdx.x + i * 256;
      const int row = e >> 6;
      const int kx  = (e & 63) + k0;
      const int g   = mt + row;
      const int n   = g / 720, l = g % 720;
      const int lh = l / 36, lw = l % 36;
      const int c  = kx / 49, rr = kx % 49;
      const int iy = lh * 3 + rr / 7 - 3;
      const int ix = lw * 3 + rr % 7 - 3;
      float v = 0.f;
      if ((unsigned)iy < (unsigned)UH && (unsigned)ix < (unsigned)UW)
        v = d2[(((size_t)n * CHN + c) * UH + iy) * UW + ix];
      As[ch & 1][row][e & 63] = v;
    }
  };
  auto issueB = [&](int ch) {
#if HAVE_TDM
    if (wave == 0)
      tdm_load_2d_f32((unsigned)(uintptr_t)&Bs[ch & 1][0][0],
                      &ss_w[(size_t)(ch * 64) * 512 + nb0],
                      /*tile*/128, 64, /*stride*/512, /*tensor*/512, CKK - ch * 64,
                      0, 0, 0);
#endif
  };

  stageA(0); issueB(0);
  for (int ch = 0; ch < NCH; ++ch) {
    if (wave == 0) wait_tensor0();
    __syncthreads();
    if (ch + 1 < NCH) { issueB(ch + 1); stageA(ch + 1); }
#pragma unroll
    for (int kc = 0; kc < 2; ++kc) {
      const int kk = kc * 32;
#if HAVE_TDM
      v16bf bfr = load_b_frag([&](int kx, int n) {
        return Bs[ch & 1][kx][wave * 16 + n];
      }, kk);
#else
      v16bf bfr = load_b_frag([&](int kx, int n) {
        return ss_w[(size_t)(ch * 64 + kx) * 512 + nbase + n];
      }, kk);
#endif
#pragma unroll
      for (int h = 0; h < 2; ++h) {
        v16bf af = load_a_frag([&](int m, int kx) { return As[ch & 1][h * 16 + m][kx]; }, kk);
        acc[h] = WMMA_BF16(af, bfr, acc[h]);
      }
    }
  }
#pragma unroll
  for (int h = 0; h < 2; ++h)
    store_d_frag(acc[h], [&](int m, int n, float v) {
      const size_t o = (size_t)(mt + h * 16 + m) * 512 + (nbase + n);
      out[o] = v + ss_b[nbase + n] + x[o];
    });
}

// ---------------------------------------------------------------------------
// Launch
// ---------------------------------------------------------------------------
extern "C" void kernel_launch(void* const* d_in, const int* in_sizes, int n_in,
                              void* d_out, int out_size, void* d_ws, size_t ws_size,
                              hipStream_t stream) {
  const float* x       = (const float*)d_in[0];
  const float* wq      = (const float*)d_in[1];
  const float* bq      = (const float*)d_in[2];
  const float* wk      = (const float*)d_in[3];
  const float* bk      = (const float*)d_in[4];
  const float* sc_w    = (const float*)d_in[5];
  const float* sc_b    = (const float*)d_in[6];
  const float* sc_bias = (const float*)d_in[7];
  const float* dec1_w  = (const float*)d_in[8];
  const float* dec1_b  = (const float*)d_in[9];
  const float* dec2_w  = (const float*)d_in[10];
  const float* dec2_b  = (const float*)d_in[11];
  const float* ss_w    = (const float*)d_in[12];
  const float* ss_b    = (const float*)d_in[13];
  float* out = (float*)d_out;

  float* ws = (float*)d_ws;
  size_t o = 0;
  float* means  = ws + o; o += (size_t)NW * CC;
  float* qbuf   = ws + o; o += (size_t)NW * CC;
  float* kbuf   = ws + o; o += (size_t)NW * CC;
  float* Sbuf   = ws + o; o += (size_t)NW * NW;
  float* scores = ws + o; o += NW;
  int*   idx    = (int*)(ws + o); o += 256;
  float* Ybuf   = ws + o; o += (size_t)5760 * CC;
  float* feat   = ws + o; o += (size_t)5760 * CKK;
  float* ffold  = ws + o; o += (size_t)BT * CHN * OH * OW;
  float* dup    = ws + o; o += (size_t)BT * CHN * UH * UW;
  float* c1     = ws + o; o += (size_t)BT * CHN * UH * UW;
  float* c2     = ws + o; o += (size_t)BT * CHN * UH * UW;
  (void)o; (void)ws_size; (void)in_sizes; (void)n_in; (void)out_size;

  // 1) window means
  k_means<<<NW, 256, 0, stream>>>(x, means);
  // 2) q / k projections (720x512 @ 512x512), pipelined async/TDM staging
  k_gemm_ws<16, CC, CC, CC, CC><<<dim3(NW / 16, CC / 128), 256, 0, stream>>>(means, wq, bq, qbuf);
  k_gemm_ws<16, CC, CC, CC, CC><<<dim3(NW / 16, CC / 128), 256, 0, stream>>>(means, wk, bk, kbuf);
  // 3) attention logits + softmax row-sums + stable top-k
  k_attn<<<dim3(NW / 16, NW / 16), 32, 0, stream>>>(qbuf, kbuf, Sbuf);
  k_scores<<<NW, 256, 0, stream>>>(Sbuf, scores);
  k_topk<<<1, 256, 0, stream>>>(scores, idx);
  // 4) gather selected windows -> Y (5760 x 512)
  k_gather<<<5760, 128, 0, stream>>>(x, idx, Ybuf);
  // 5) GEMM1: feat = Y @ sc_w + sc_b  (5760 x 6272)
  k_gemm_ws<32, CC, CC, CKK, CKK><<<dim3(5760 / 32, CKK / 128), 256, 0, stream>>>(Ybuf, sc_w, sc_b, feat);
  // 6) fold + sc_bias
  k_fold<<<(BT * CHN * OH * OW + 255) / 256, 256, 0, stream>>>(feat, sc_bias, ffold);
  // 7) bilinear x2
  k_up2<<<(BT * CHN * UH * UW + 255) / 256, 256, 0, stream>>>(ffold, dup);
  // 8) conv1 + leaky, conv2
  k_conv3<<<(BT * UH * UW) / 32, 256, 0, stream>>>(dup, dec1_w, dec1_b, c1, 1);
  k_conv3<<<(BT * UH * UW) / 32, 256, 0, stream>>>(c1, dec2_w, dec2_b, c2, 0);
  // 9) implicit unfold + GEMM2 + residual -> out
  k_gemm2<<<dim3((BT * 720) / 32, CC / 128), 256, 0, stream>>>(c2, ss_w, ss_b, x, out);
}
